// Piston_PolicyHelper_51445118271693
// MI455X (gfx1250) — matrix operand order repaired and checked by reading, not compiled, not run on hardware
//
#include <hip/hip_runtime.h>
#include <hip/hip_bf16.h>
#include <math.h>

// ---------------------------------------------------------------------------
// Sizes (match reference)
// ---------------------------------------------------------------------------
#define OBS_  4096
#define ENC_  1024
#define LAT_  512
#define ACT_  3
#define BATCH 4096
#define NMAX_ 8

typedef __attribute__((ext_vector_type(16))) __bf16 v16bf;
typedef __attribute__((ext_vector_type(8)))  __bf16 v8bf;
typedef __attribute__((ext_vector_type(4)))  __bf16 v4bf;
typedef __attribute__((ext_vector_type(8)))  float  v8f;
typedef __attribute__((ext_vector_type(4)))  int    v4i;

typedef __attribute__((address_space(1))) v4i gas_v4i;   // global (HIP __device__)
typedef __attribute__((address_space(3))) v4i las_v4i;   // LDS

#define HAS_ASYNC_LDS __has_builtin(__builtin_amdgcn_global_load_async_to_lds_b128)

// fp32 -> bf16 round-to-nearest-even (prepass / epilogues only)
__device__ __forceinline__ __bf16 f2bf(float f) {
    unsigned u = __builtin_bit_cast(unsigned, f);
    unsigned r = u + 0x7FFFu + ((u >> 16) & 1u);
    unsigned short h = (unsigned short)(r >> 16);
    return __builtin_bit_cast(__bf16, h);
}

// 16-byte Global -> LDS copy: async engine if available, else direct store.
__device__ __forceinline__ void copy16_g2l(const __bf16* g, __bf16* l) {
#if HAS_ASYNC_LDS
    __builtin_amdgcn_global_load_async_to_lds_b128((gas_v4i*)g, (las_v4i*)l, 0, 0);
#else
    *(v8bf*)l = *(const v8bf*)g;
#endif
}

__device__ __forceinline__ void async_wait_all() {
#if HAS_ASYNC_LDS
#if __has_builtin(__builtin_amdgcn_s_wait_asynccnt)
    __builtin_amdgcn_s_wait_asynccnt(0);
#else
    asm volatile("s_wait_asynccnt 0x0" ::: "memory");
#endif
#endif
}

// ---------------------------------------------------------------------------
// Prepass: fp32 -> bf16 elementwise, 4 elements/thread (all sizes %4 == 0)
// ---------------------------------------------------------------------------
__global__ void __launch_bounds__(256)
cvt_f32_to_bf16(const float* __restrict__ in, __bf16* __restrict__ out, int n4) {
    int i = blockIdx.x * blockDim.x + threadIdx.x;
    if (i < n4) {
        float4 f = ((const float4*)in)[i];
        v4bf o;
        o[0] = f2bf(f.x); o[1] = f2bf(f.y); o[2] = f2bf(f.z); o[3] = f2bf(f.w);
        ((v4bf*)out)[i] = o;
    }
}

// ---------------------------------------------------------------------------
// WMMA bf16 fragment helpers (16x16x32, wave32)
// A fragment, lane L: row = (L&15), half = L>>4:
//   elems 0..7  at [row][kofs + half*8 .. +7], elems 8..15 at +16.
// B fragment from row-major W[N][K], lane L: col = n0+(L&15):
//   elems 0..7 = W[col][kbase + half*8 .. +7], elems 8..15 at +16.
// ---------------------------------------------------------------------------
__device__ __forceinline__ v16bf frag_from_rowmajor(const __bf16* base, int ld,
                                                    int row, int lane) {
    int half = lane >> 4;
    const __bf16* p = base + (size_t)row * ld + half * 8;
    v8bf lo = *(const v8bf*)(p);
    v8bf hi = *(const v8bf*)(p + 16);
    return __builtin_shufflevector(lo, hi, 0, 1, 2, 3, 4, 5, 6, 7,
                                           8, 9, 10, 11, 12, 13, 14, 15);
}

__device__ __forceinline__ v16bf load_B_bf(const __bf16* __restrict__ W,
                                           int ldw, int n0, int kbase, int lane) {
    int col  = n0 + (lane & 15);
    int half = lane >> 4;
    const __bf16* p = W + (size_t)col * ldw + kbase + half * 8;
    v8bf lo = *(const v8bf*)(p);
    v8bf hi = *(const v8bf*)(p + 16);
    return __builtin_shufflevector(lo, hi, 0, 1, 2, 3, 4, 5, 6, 7,
                                           8, 9, 10, 11, 12, 13, 14, 15);
}

__device__ __forceinline__ v8f wmma_bf16(v16bf a, v16bf b, v8f c) {
    return __builtin_amdgcn_wmma_f32_16x16x32_bf16(false, a, false, b,
                                                   (short)0, c, false, false);
}

// ---------------------------------------------------------------------------
// Kernel 1: act(A[M,K] @ W[N,K]^T + bias[N]) -> optional f32 C / bf16 Cb
// Block = 8 waves, tile BM=32 x BN=256; wave owns 32x32 (4 accumulators).
// A panel (32x32 bf16, shared by all waves) staged via async Global->LDS,
// double-buffered; B fragments streamed from global (hit L2: 192 MB).
// grid: (M/32, N/256)
// ---------------------------------------------------------------------------
__global__ void __launch_bounds__(256)
gemm_bias_act(const __bf16* __restrict__ A, int lda,
              const __bf16* __restrict__ W, int ldw,
              const float* __restrict__ bias,
              float* __restrict__ C, __bf16* __restrict__ Cb,
              int ldc, int K, int act) {
    __shared__ __bf16 sA[2][32 * 32];          // 2 x 2 KB

    int tid  = threadIdx.x;
    int lane = tid & 31;
    int wave = tid >> 5;
    int m0 = blockIdx.x * 32;
    int n0 = blockIdx.y * 256 + wave * 32;

    // stage A panel for k-chunk: 128 chunks of 16B, threads 0..127
    int srow = tid >> 2, sc = tid & 3;         // row 0..31, 16B-chunk 0..3
    const __bf16* gA = A + (size_t)(m0 + srow) * lda + sc * 8;

    if (tid < 128) copy16_g2l(gA, &sA[0][srow * 32 + sc * 8]);
    async_wait_all();
    __syncthreads();

    v8f acc00 = {}, acc01 = {}, acc10 = {}, acc11 = {};

    for (int k = 0; k < K; k += 32) {
        int cur = (k >> 5) & 1;
        // prefetch next A panel into the other buffer
        if (k + 32 < K && tid < 128)
            copy16_g2l(gA + k + 32, &sA[cur ^ 1][srow * 32 + sc * 8]);

        v16bf a0 = frag_from_rowmajor(sA[cur], 32, (lane & 15), lane);
        v16bf a1 = frag_from_rowmajor(sA[cur], 32, 16 + (lane & 15), lane);
        v16bf b0 = load_B_bf(W, ldw, n0,      k, lane);
        v16bf b1 = load_B_bf(W, ldw, n0 + 16, k, lane);

        acc00 = wmma_bf16(a0, b0, acc00);
        acc01 = wmma_bf16(a0, b1, acc01);
        acc10 = wmma_bf16(a1, b0, acc10);
        acc11 = wmma_bf16(a1, b1, acc11);

        async_wait_all();
        __syncthreads();
    }

    // epilogue: 4 tiles; C/D layout: col = n + (lane&15), row = m + half*8 + i
    int half = lane >> 4;
#pragma unroll
    for (int ms = 0; ms < 2; ++ms) {
#pragma unroll
        for (int ns = 0; ns < 2; ++ns) {
            v8f acc = (ms == 0) ? (ns == 0 ? acc00 : acc01)
                                : (ns == 0 ? acc10 : acc11);
            int col   = n0 + ns * 16 + (lane & 15);
            int rbase = m0 + ms * 16 + half * 8;
            float bv  = bias[col];
#pragma unroll
            for (int i = 0; i < 8; ++i) {
                float v = acc[i] + bv;
                if (act == 1) v = fmaxf(v, 0.0f);
                if (C)  C[(size_t)(rbase + i) * ldc + col]  = v;
                if (Cb) Cb[(size_t)(rbase + i) * ldc + col] = f2bf(v);
            }
        }
    }
}

// ---------------------------------------------------------------------------
// Kernel 2: state_vals[row] = tanh(enc[row,:] . W_v + b_v)  (wave per row)
// ---------------------------------------------------------------------------
__global__ void __launch_bounds__(256)
vnet_kernel(const __bf16* __restrict__ enc, const float* __restrict__ Wv,
            const float* __restrict__ bv, float* __restrict__ out) {
    int row  = blockIdx.x * 8 + (threadIdx.x >> 5);
    int lane = threadIdx.x & 31;
    const __bf16* e = enc + (size_t)row * ENC_;
    float s = 0.0f;
    for (int k = lane; k < ENC_; k += 32) s += (float)e[k] * Wv[k];
#pragma unroll
    for (int off = 16; off; off >>= 1) s += __shfl_xor(s, off, 32);
    if (lane == 0) out[row] = tanhf(s + bv[0]);
}

// ---------------------------------------------------------------------------
// Kernel 3: one GRU step.  Block = 16 batch rows x 128 h-cols (8 waves).
// The two shared A panels (x_t and h, 16x32 bf16 each) are staged via async
// Global->LDS with double buffering; each wave runs 6 WMMA chains (r/z/n for
// x-path and h-path).  Epilogue: gates + ragged mask; h ping-pongs fp32+bf16.
// grid: (B/16, LAT/128)
// ---------------------------------------------------------------------------
__global__ void __launch_bounds__(256)
gru_step(const __bf16* __restrict__ Xb,        // neighbors bf16 [B, NMAX, LAT]
         int t,
         const float*  __restrict__ h_in,      // fp32 [B, LAT]
         const __bf16* __restrict__ h_in_bf,   // bf16 [B, LAT]
         const __bf16* __restrict__ W_ih,      // bf16 [3L, L]
         const float*  __restrict__ b_ih,
         const __bf16* __restrict__ W_hh,      // bf16 [3L, L]
         const float*  __restrict__ b_hh,
         const int*    __restrict__ counts,    // [B]
         float*        __restrict__ h_out,     // fp32 [B, LAT]
         __bf16*       __restrict__ h_out_bf)  // bf16 [B, LAT]
{
    __shared__ __bf16 sP[2][2 * 16 * 32];      // [buf][ X(16x32) | H(16x32) ]

    int tid  = threadIdx.x;
    int lane = tid & 31;
    int wave = tid >> 5;
    int m0 = blockIdx.x * 16;
    int n0 = blockIdx.y * 128 + wave * 16;

    // staging: 128 chunks of 16B (64 for X, 64 for H); threads 0..127
    int pan = (tid >> 6) & 1;                  // 0 = X, 1 = H
    int srow = (tid >> 2) & 15, sc = tid & 3;
    const __bf16* gX = Xb + ((size_t)(m0 + srow) * NMAX_ + t) * LAT_ + sc * 8;
    const __bf16* gH = h_in_bf + (size_t)(m0 + srow) * LAT_ + sc * 8;
    const __bf16* gsrc = pan ? gH : gX;
    int ldst = pan * (16 * 32) + srow * 32 + sc * 8;

    if (tid < 128) copy16_g2l(gsrc, &sP[0][ldst]);
    async_wait_all();
    __syncthreads();

    v8f air = {}, aiz = {}, ain = {};
    v8f ahr = {}, ahz = {}, ahn = {};

    for (int k = 0; k < LAT_; k += 32) {
        int cur = (k >> 5) & 1;
        if (k + 32 < LAT_ && tid < 128)
            copy16_g2l(gsrc + k + 32, &sP[cur ^ 1][ldst]);

        v16bf ax = frag_from_rowmajor(sP[cur], 32, (lane & 15), lane);
        v16bf ah = frag_from_rowmajor(sP[cur] + 16 * 32, 32, (lane & 15), lane);

        v16bf b0 = load_B_bf(W_ih, LAT_, n0,            k, lane);
        v16bf b1 = load_B_bf(W_ih, LAT_, n0 + LAT_,     k, lane);
        v16bf b2 = load_B_bf(W_ih, LAT_, n0 + 2 * LAT_, k, lane);
        air = wmma_bf16(ax, b0, air);
        aiz = wmma_bf16(ax, b1, aiz);
        ain = wmma_bf16(ax, b2, ain);

        b0 = load_B_bf(W_hh, LAT_, n0,            k, lane);
        b1 = load_B_bf(W_hh, LAT_, n0 + LAT_,     k, lane);
        b2 = load_B_bf(W_hh, LAT_, n0 + 2 * LAT_, k, lane);
        ahr = wmma_bf16(ah, b0, ahr);
        ahz = wmma_bf16(ah, b1, ahz);
        ahn = wmma_bf16(ah, b2, ahn);

        async_wait_all();
        __syncthreads();
    }

    int col   = n0 + (lane & 15);
    int rbase = m0 + (lane >> 4) * 8;
    float bir = b_ih[col], biz = b_ih[col + LAT_], bin_ = b_ih[col + 2 * LAT_];
    float bhr = b_hh[col], bhz = b_hh[col + LAT_], bhn  = b_hh[col + 2 * LAT_];

#pragma unroll
    for (int i = 0; i < 8; ++i) {
        int row = rbase + i;
        float hprev = h_in[(size_t)row * LAT_ + col];
        float r = 1.0f / (1.0f + __expf(-(air[i] + bir + ahr[i] + bhr)));
        float z = 1.0f / (1.0f + __expf(-(aiz[i] + biz + ahz[i] + bhz)));
        float n = tanhf(ain[i] + bin_ + r * (ahn[i] + bhn));
        float hn = (1.0f - z) * n + z * hprev;
        bool valid = t < counts[row];
        float hw = valid ? hn : hprev;
        h_out[(size_t)row * LAT_ + col]    = hw;
        h_out_bf[(size_t)row * LAT_ + col] = f2bf(hw);
    }
}

// ---------------------------------------------------------------------------
// Kernel 4: probs = softmax(relu(h @ W_pr^T + b_pr))  (ACT=3, wave per row)
// ---------------------------------------------------------------------------
__global__ void __launch_bounds__(256)
probs_kernel(const float* __restrict__ h, const float* __restrict__ Wpr,
             const float* __restrict__ bpr, float* __restrict__ out) {
    int row  = blockIdx.x * 8 + (threadIdx.x >> 5);
    int lane = threadIdx.x & 31;
    const float* hp = h + (size_t)row * LAT_;
    float s0 = 0.f, s1 = 0.f, s2 = 0.f;
    for (int k = lane; k < LAT_; k += 32) {
        float hv = hp[k];
        s0 += hv * Wpr[k];
        s1 += hv * Wpr[LAT_ + k];
        s2 += hv * Wpr[2 * LAT_ + k];
    }
#pragma unroll
    for (int off = 16; off; off >>= 1) {
        s0 += __shfl_xor(s0, off, 32);
        s1 += __shfl_xor(s1, off, 32);
        s2 += __shfl_xor(s2, off, 32);
    }
    if (lane == 0) {
        float l0 = fmaxf(s0 + bpr[0], 0.f);
        float l1 = fmaxf(s1 + bpr[1], 0.f);
        float l2 = fmaxf(s2 + bpr[2], 0.f);
        float m  = fmaxf(l0, fmaxf(l1, l2));
        float e0 = __expf(l0 - m), e1 = __expf(l1 - m), e2 = __expf(l2 - m);
        float inv = 1.0f / (e0 + e1 + e2);
        out[(size_t)row * ACT_ + 0] = e0 * inv;
        out[(size_t)row * ACT_ + 1] = e1 * inv;
        out[(size_t)row * ACT_ + 2] = e2 * inv;
    }
}

// ---------------------------------------------------------------------------
// Host-side launcher
// ---------------------------------------------------------------------------
static inline void cvt_launch(const float* in, __bf16* out, size_t n,
                              hipStream_t stream) {
    int n4 = (int)(n / 4);
    cvt_f32_to_bf16<<<dim3((n4 + 255) / 256), dim3(256), 0, stream>>>(in, out, n4);
}

extern "C" void kernel_launch(void* const* d_in, const int* in_sizes, int n_in,
                              void* d_out, int out_size, void* d_ws, size_t ws_size,
                              hipStream_t stream) {
    const float* observation = (const float*)d_in[0];   // [B, OBS]
    const float* neighbors   = (const float*)d_in[1];   // [B, NMAX, LAT]
    const int*   counts      = (const int*)  d_in[2];   // [B]
    const float* W_ds  = (const float*)d_in[3];         // [ENC, OBS]
    const float* b_ds  = (const float*)d_in[4];
    const float* W_pol = (const float*)d_in[5];         // [LAT, ENC]
    const float* b_pol = (const float*)d_in[6];
    const float* W_v   = (const float*)d_in[7];         // [1, ENC]
    const float* b_v   = (const float*)d_in[8];
    const float* W_ih  = (const float*)d_in[9];         // [3L, L]
    const float* b_ih  = (const float*)d_in[10];
    const float* W_hh  = (const float*)d_in[11];        // [3L, L]
    const float* b_hh  = (const float*)d_in[12];
    const float* W_pr  = (const float*)d_in[13];        // [ACT, LAT]
    const float* b_pr  = (const float*)d_in[14];

    float* probs_out = (float*)d_out;                          // [B, ACT]
    float* vals_out  = (float*)d_out + (size_t)BATCH * ACT_;   // [B, 1]

    // ----- workspace carve-up -----
    char* ws = (char*)d_ws;
    __bf16* obs_b  = (__bf16*)ws;  ws += (size_t)BATCH * OBS_ * 2;        // 32 MB
    __bf16* nei_b  = (__bf16*)ws;  ws += (size_t)BATCH * NMAX_ * LAT_ * 2;// 32 MB
    __bf16* Wds_b  = (__bf16*)ws;  ws += (size_t)ENC_ * OBS_ * 2;         //  8 MB
    __bf16* Wpol_b = (__bf16*)ws;  ws += (size_t)LAT_ * ENC_ * 2;         //  1 MB
    __bf16* Wih_b  = (__bf16*)ws;  ws += (size_t)3 * LAT_ * LAT_ * 2;     // 1.5 MB
    __bf16* Whh_b  = (__bf16*)ws;  ws += (size_t)3 * LAT_ * LAT_ * 2;     // 1.5 MB
    __bf16* enc_b  = (__bf16*)ws;  ws += (size_t)BATCH * ENC_ * 2;        //  8 MB
    float*  h0     = (float*)ws;   ws += (size_t)BATCH * LAT_ * 4;        //  8 MB
    float*  h1     = (float*)ws;   ws += (size_t)BATCH * LAT_ * 4;        //  8 MB
    __bf16* h0_b   = (__bf16*)ws;  ws += (size_t)BATCH * LAT_ * 2;        //  4 MB
    __bf16* h1_b   = (__bf16*)ws;  ws += (size_t)BATCH * LAT_ * 2;        //  4 MB

    // ----- prepass: fp32 -> bf16 for all WMMA operands -----
    cvt_launch(observation, obs_b, (size_t)BATCH * OBS_,         stream);
    cvt_launch(neighbors,   nei_b, (size_t)BATCH * NMAX_ * LAT_, stream);
    cvt_launch(W_ds,  Wds_b,  (size_t)ENC_ * OBS_,     stream);
    cvt_launch(W_pol, Wpol_b, (size_t)LAT_ * ENC_,     stream);
    cvt_launch(W_ih,  Wih_b,  (size_t)3 * LAT_ * LAT_, stream);
    cvt_launch(W_hh,  Whh_b,  (size_t)3 * LAT_ * LAT_, stream);

    dim3 blk(256);

    // enc = relu(obs @ W_ds^T + b_ds)  -> bf16 only
    gemm_bias_act<<<dim3(BATCH / 32, ENC_ / 256), blk, 0, stream>>>(
        obs_b, OBS_, Wds_b, OBS_, b_ds, nullptr, enc_b, ENC_, OBS_, 1);

    // h0 = pol = relu(enc @ W_pol^T + b_pol)  -> fp32 + bf16
    gemm_bias_act<<<dim3(BATCH / 32, LAT_ / 256), blk, 0, stream>>>(
        enc_b, ENC_, Wpol_b, ENC_, b_pol, h0, h0_b, LAT_, ENC_, 1);

    // state_vals = tanh(enc @ W_v^T + b_v)
    vnet_kernel<<<dim3(BATCH / 8), blk, 0, stream>>>(enc_b, W_v, b_v, vals_out);

    // 8 GRU steps, ping-pong; final h lands in h0 (even step count)
    for (int t = 0; t < NMAX_; ++t) {
        const float*  hi  = (t & 1) ? h1   : h0;
        const __bf16* hib = (t & 1) ? h1_b : h0_b;
        float*        ho  = (t & 1) ? h0   : h1;
        __bf16*       hob = (t & 1) ? h0_b : h1_b;
        gru_step<<<dim3(BATCH / 16, LAT_ / 128), blk, 0, stream>>>(
            nei_b, t, hi, hib, Wih_b, b_ih, Whh_b, b_hh, counts, ho, hob);
    }

    // probs = softmax(relu(h @ W_pr^T + b_pr))
    probs_kernel<<<dim3(BATCH / 8), blk, 0, stream>>>(h0, W_pr, b_pr, probs_out);
}
